// TinySAModule_14525579395845
// MI455X (gfx1250) — matrix operand order repaired and checked by reading, not compiled, not run on hardware
//
#include <hip/hip_runtime.h>

// ---------------------------------------------------------------------------
// TinySAModule on MI455X (gfx1250): WMMA f16 GEMMs with pre-packed B frags
// ---------------------------------------------------------------------------
#define TSA_DIM   128
#define TSA_H     4
#define TSA_DH    32
#define TSA_MAXK  32
#define TSA_KNN   8
#define TSA_R2    0.09f                // RADIUS^2 (0.3^2)
#define TSA_SCALE 0.17677669529663689f // DH^-0.5
#define TSA_CAND  2048                 // LDS candidate buffer per center
#define TSA_MAXM  2560                 // max centers staged in LDS for KNN

typedef __attribute__((ext_vector_type(16))) _Float16 v16h;
typedef __attribute__((ext_vector_type(8)))  float    v8f;

// ---------------- WMMA A-fragment loader (f32 memory -> f16 frag) ----------
// A (16x32, f16): lane l<16 -> row l, K = {ks..ks+7, ks+16..ks+23}, ks=k0+hi*8
__device__ __forceinline__ v16h tsa_load_a(const float* __restrict__ A, int lda,
                                           int row, int ks, bool valid) {
  v16h a;
  if (valid) {
    const float* p = A + (size_t)row * lda + ks;
#pragma unroll
    for (int i = 0; i < 8; ++i) a[i] = (_Float16)p[i];
#pragma unroll
    for (int i = 0; i < 8; ++i) a[8 + i] = (_Float16)p[16 + i];
  } else {
#pragma unroll
    for (int i = 0; i < 16; ++i) a[i] = (_Float16)0.0f;
  }
  return a;
}

// ---------------- Weight pre-pack: f32 row-major -> f16 B fragments --------
// Frag index t = (tn*ktiles + kt)*32 + lane ; element i = B[(kb+i+(i>=8?8:0))*ncols + c]
// with c = tn*16 + (lane&15), kb = kt*32 + (lane>>4)*8.
__global__ void tsa_pack_b(const float* __restrict__ B, int kdim, int ncols,
                           _Float16* __restrict__ Bp) {
  const int ktiles = kdim >> 5, ntiles = ncols >> 4;
  const int total = ntiles * ktiles * 32;
  const int t = blockIdx.x * blockDim.x + threadIdx.x;
  if (t >= total) return;
  const int lane = t & 31;
  const int kt = (t >> 5) % ktiles;
  const int tn = (t >> 5) / ktiles;
  const int l = lane & 15, hi = lane >> 4;
  const int c = tn * 16 + l;
  const int kb = kt * 32 + hi * 8;
  _Float16* dst = Bp + (size_t)t * 16;
#pragma unroll
  for (int i = 0; i < 16; ++i)
    dst[i] = (_Float16)B[(size_t)(kb + i + (i >= 8 ? 8 : 0)) * ncols + c];
}

__device__ __forceinline__ v16h tsa_load_bp(const _Float16* __restrict__ Bp,
                                            int ktiles, int tn, int kt,
                                            int lane) {
  return ((const v16h*)Bp)[(size_t)(tn * ktiles + kt) * 32 + lane];
}

// ---------------- Kernel 1: ball query (top-32 within radius) --------------
__global__ void tsa_ballquery(const float* __restrict__ xyz,
                              const int* __restrict__ idxc,
                              int N, int* __restrict__ nbr) {
  __shared__ float cd[TSA_CAND];
  __shared__ int   ci[TSA_CAND];
  __shared__ int   cnt;
  const int m = blockIdx.x;
  const int ctr = idxc[m];
  const float cx = xyz[ctr * 3 + 0], cy = xyz[ctr * 3 + 1], cz = xyz[ctr * 3 + 2];
  if (threadIdx.x == 0) cnt = 0;
  __syncthreads();
  for (int n = threadIdx.x; n < N; n += blockDim.x) {
    float dx = xyz[n * 3 + 0] - cx;
    float dy = xyz[n * 3 + 1] - cy;
    float dz = xyz[n * 3 + 2] - cz;
    float d2 = dx * dx + dy * dy + dz * dz;
    if (d2 < TSA_R2) {
      int p = atomicAdd(&cnt, 1);
      if (p < TSA_CAND) { cd[p] = d2; ci[p] = n; }
    }
  }
  __syncthreads();
  const int C = min(cnt, TSA_CAND);
  if (C <= TSA_MAXK) {
    for (int j = threadIdx.x; j < TSA_MAXK; j += blockDim.x)
      nbr[(size_t)m * TSA_MAXK + j] = (j < C) ? ci[j] : -1;
  } else {
    // rank-select: candidate goes to slot == its rank (set semantics suffice;
    // attention/softmax are permutation-invariant over the neighbor axis)
    for (int p = threadIdx.x; p < C; p += blockDim.x) {
      float dp = cd[p]; int ip = ci[p];
      int rank = 0;
      for (int q = 0; q < C; ++q) {
        float dq = cd[q];
        rank += (dq < dp) || (dq == dp && ci[q] < ip);
      }
      if (rank < TSA_MAXK) nbr[(size_t)m * TSA_MAXK + rank] = ip;
    }
  }
}

// ---------------- Kernel 2: q = gather(feats, idxc) @ Wq -------------------
__global__ void tsa_proj_center(const float* __restrict__ feats,
                                const int* __restrict__ idxc, int M,
                                const _Float16* __restrict__ Wqp,
                                float* __restrict__ qm) {
  const int gw = (int)(blockIdx.x * blockDim.x + threadIdx.x) >> 5;
  const int ntile = TSA_DIM / 16, ktiles = TSA_DIM / 32;
  const int tiles_m = (M + 15) >> 4;
  const int tm = gw / ntile, tn = gw - tm * ntile;
  if (tm >= tiles_m) return;
  const int lane = threadIdx.x & 31, l = lane & 15, hi = lane >> 4;
  const int row = tm * 16 + l;
  const bool av = row < M;
  const int src = av ? idxc[row] : 0;
  const int n0 = tn << 4;
  v8f acc = {};
  for (int kt = 0; kt < ktiles; ++kt) {
    v16h a = tsa_load_a(feats, TSA_DIM, src, kt * 32 + hi * 8, av);
    v16h b = tsa_load_bp(Wqp, ktiles, tn, kt, lane);
    acc = __builtin_amdgcn_wmma_f32_16x16x32_f16(false, a, false, b, (short)0,
                                                 acc, false, false);
  }
#pragma unroll
  for (int i = 0; i < 8; ++i) {
    int r = tm * 16 + i + hi * 8;
    if (r < M) qm[(size_t)r * TSA_DIM + n0 + l] = acc[i];
  }
}

// ---------------- Kernel 3: K/V = gather(feats, nbr) @ Wk / Wv -------------
__global__ void tsa_proj_kv(const float* __restrict__ feats,
                            const int* __restrict__ nbr, int rowsKV,
                            const _Float16* __restrict__ Wkp,
                            const _Float16* __restrict__ Wvp,
                            _Float16* __restrict__ km,
                            _Float16* __restrict__ vm) {
  const int gw = (int)(blockIdx.x * blockDim.x + threadIdx.x) >> 5;
  const int ntile = TSA_DIM / 16, ktiles = TSA_DIM / 32;
  const int tiles_m = (rowsKV + 15) >> 4;
  const int tm = gw / ntile, tn = gw - tm * ntile;
  if (tm >= tiles_m) return;
  const int lane = threadIdx.x & 31, l = lane & 15, hi = lane >> 4;
  const int row = tm * 16 + l;               // rowsKV = M*32, multiple of 16
  const int src = nbr[row];
  const bool av = src >= 0;
  const int n0 = tn << 4;
  v8f accK = {}, accV = {};
  for (int kt = 0; kt < ktiles; ++kt) {
    v16h a  = tsa_load_a(feats, TSA_DIM, av ? src : 0, kt * 32 + hi * 8, av);
    v16h bk = tsa_load_bp(Wkp, ktiles, tn, kt, lane);
    v16h bv = tsa_load_bp(Wvp, ktiles, tn, kt, lane);
    accK = __builtin_amdgcn_wmma_f32_16x16x32_f16(false, a, false, bk, (short)0,
                                                  accK, false, false);
    accV = __builtin_amdgcn_wmma_f32_16x16x32_f16(false, a, false, bv, (short)0,
                                                  accV, false, false);
  }
#pragma unroll
  for (int i = 0; i < 8; ++i) {
    int r = tm * 16 + i + hi * 8;
    size_t o = (size_t)r * TSA_DIM + n0 + l;
    km[o] = (_Float16)accK[i];
    vm[o] = (_Float16)accV[i];
  }
}

// ---------------- Kernel 4: masked local attention (tiny heads) ------------
__global__ void tsa_attn(const float* __restrict__ qm,
                         const _Float16* __restrict__ km,
                         const _Float16* __restrict__ vm,
                         const int* __restrict__ nbr,
                         float* __restrict__ outm) {
  __shared__ float sq[TSA_DIM];
  __shared__ float sattn[TSA_H][TSA_MAXK];
  const int m = blockIdx.x;
  const int tid = threadIdx.x;                 // 128 threads = 4 waves
  sq[tid] = qm[(size_t)m * TSA_DIM + tid];
  __syncthreads();
  const int h = tid >> 5, j = tid & 31;
  const bool valid = nbr[(size_t)m * TSA_MAXK + j] >= 0;
  float logit = -1e9f;
  if (valid) {
    const _Float16* kp = km + ((size_t)m * TSA_MAXK + j) * TSA_DIM + h * TSA_DH;
    float s = 0.0f;
#pragma unroll
    for (int d = 0; d < TSA_DH; ++d) s += sq[h * TSA_DH + d] * (float)kp[d];
    logit = s * TSA_SCALE;
  }
  // wave32 softmax over the 32 neighbors of this head
  float mx = logit;
#pragma unroll
  for (int o = 16; o > 0; o >>= 1) mx = fmaxf(mx, __shfl_xor(mx, o, 32));
  float e = __expf(logit - mx);
  float sum = e;
#pragma unroll
  for (int o = 16; o > 0; o >>= 1) sum += __shfl_xor(sum, o, 32);
  sattn[h][j] = e / sum;
  __syncthreads();
  // re-map threads: (h, d) computes sum_j attn * v  ; store (D,H)-flattened
  const int d = j;
  float acc = 0.0f;
#pragma unroll
  for (int jj = 0; jj < TSA_MAXK; ++jj)
    acc += sattn[h][jj] *
           (float)vm[((size_t)m * TSA_MAXK + jj) * TSA_DIM + h * TSA_DH + d];
  outm[(size_t)m * TSA_DIM + d * TSA_H + h] = acc;  // torch-faithful (D,H)
}

// ---------------- Generic row-major WMMA GEMM with epilogue ----------------
// mode 0: C = acc + bias ; mode 1: C = relu(acc + bias)
__global__ void tsa_gemm_rm(const float* __restrict__ A, int rows, int kdim,
                            const _Float16* __restrict__ Bp, int ncols,
                            const float* __restrict__ bias, int mode,
                            float* __restrict__ C) {
  const int gw = (int)(blockIdx.x * blockDim.x + threadIdx.x) >> 5;
  const int ntile = ncols >> 4;
  const int ktiles = kdim >> 5;
  const int tiles_m = (rows + 15) >> 4;
  const int tm = gw / ntile, tn = gw - tm * ntile;
  if (tm >= tiles_m) return;
  const int lane = threadIdx.x & 31, l = lane & 15, hi = lane >> 4;
  const int row = tm * 16 + l;
  const bool av = row < rows;
  const int n0 = tn << 4;
  v8f acc = {};
  for (int kt = 0; kt < ktiles; ++kt) {
    v16h a = tsa_load_a(A, kdim, av ? row : 0, kt * 32 + hi * 8, av);
    v16h b = tsa_load_bp(Bp, ktiles, tn, kt, lane);
    acc = __builtin_amdgcn_wmma_f32_16x16x32_f16(false, a, false, b, (short)0,
                                                 acc, false, false);
  }
  const float bb = bias ? bias[n0 + l] : 0.0f;
#pragma unroll
  for (int i = 0; i < 8; ++i) {
    int r = tm * 16 + i + hi * 8;
    if (r < rows) {
      float v = acc[i] + bb;
      if (mode == 1) v = fmaxf(v, 0.0f);
      C[(size_t)r * ncols + n0 + l] = v;
    }
  }
}

// ---------------- LayerNorm + residual: out = base[gidx] + LN(x)*w + b -----
__global__ void tsa_ln(const float* __restrict__ x,
                       const float* __restrict__ basemat,
                       const int* __restrict__ gidx,
                       const float* __restrict__ w, const float* __restrict__ b,
                       float* __restrict__ out) {
  __shared__ float red[TSA_DIM];
  const int m = blockIdx.x, c = threadIdx.x;   // 128 threads
  const float xv = x[(size_t)m * TSA_DIM + c];
  red[c] = xv;
  __syncthreads();
#pragma unroll
  for (int s = TSA_DIM / 2; s > 0; s >>= 1) {
    if (c < s) red[c] += red[c + s];
    __syncthreads();
  }
  const float mu = red[0] * (1.0f / TSA_DIM);
  __syncthreads();
  const float d = xv - mu;
  red[c] = d * d;
  __syncthreads();
#pragma unroll
  for (int s = TSA_DIM / 2; s > 0; s >>= 1) {
    if (c < s) red[c] += red[c + s];
    __syncthreads();
  }
  const float var = red[0] * (1.0f / TSA_DIM);
  const int br = gidx ? gidx[m] : m;
  out[(size_t)m * TSA_DIM + c] =
      basemat[(size_t)br * TSA_DIM + c] + d * rsqrtf(var + 1e-5f) * w[c] + b[c];
}

// ---------------- KNN top-8 centers per point + IDW weights ----------------
__global__ void tsa_knn(const float* __restrict__ xyz,
                        const int* __restrict__ idxc, int N, int M,
                        int* __restrict__ kidx, float* __restrict__ kw) {
  __shared__ float scx[TSA_MAXM * 3];
  const int Mc = min(M, TSA_MAXM);
  for (int mm = threadIdx.x; mm < Mc; mm += blockDim.x) {
    int c = idxc[mm];
    scx[mm * 3 + 0] = xyz[c * 3 + 0];
    scx[mm * 3 + 1] = xyz[c * 3 + 1];
    scx[mm * 3 + 2] = xyz[c * 3 + 2];
  }
  __syncthreads();
  const int n = blockIdx.x * blockDim.x + threadIdx.x;
  if (n >= N) return;
  const float px = xyz[n * 3 + 0], py = xyz[n * 3 + 1], pz = xyz[n * 3 + 2];
  float bd[TSA_KNN]; int bi[TSA_KNN];
#pragma unroll
  for (int t = 0; t < TSA_KNN; ++t) { bd[t] = 3.4e38f; bi[t] = 0; }
  for (int mm = 0; mm < Mc; ++mm) {
    float dx = scx[mm * 3 + 0] - px;
    float dy = scx[mm * 3 + 1] - py;
    float dz = scx[mm * 3 + 2] - pz;
    float d2 = dx * dx + dy * dy + dz * dz;
    if (d2 < bd[TSA_KNN - 1]) {           // stable insert (ties keep lower m)
      bd[TSA_KNN - 1] = d2; bi[TSA_KNN - 1] = mm;
#pragma unroll
      for (int p = TSA_KNN - 1; p > 0; --p) {
        if (bd[p] < bd[p - 1]) {
          float td = bd[p]; bd[p] = bd[p - 1]; bd[p - 1] = td;
          int   ti = bi[p]; bi[p] = bi[p - 1]; bi[p - 1] = ti;
        }
      }
    }
  }
  float wv[TSA_KNN]; float sum = 0.0f;
#pragma unroll
  for (int t = 0; t < TSA_KNN; ++t) {
    float dist = sqrtf(bd[t]);
    float u = dist + 1e-6f;
    wv[t] = 1.0f / (u * u);               // ALPHA = 2
    sum += wv[t];
  }
  const float inv = 1.0f / sum;
#pragma unroll
  for (int t = 0; t < TSA_KNN; ++t) {
    kidx[(size_t)n * TSA_KNN + t] = bi[t];
    kw[(size_t)n * TSA_KNN + t] = wv[t] * inv;
  }
}

// ---------------- IDW upsample: one wave per point -------------------------
__global__ void tsa_upsample(const float* __restrict__ cf2,
                             const int* __restrict__ kidx,
                             const float* __restrict__ kw, int N,
                             float* __restrict__ of) {
  const int wid = (int)(blockIdx.x * blockDim.x + threadIdx.x) >> 5;
  const int lane = threadIdx.x & 31;
  if (wid >= N) return;
  float acc[4] = {0.f, 0.f, 0.f, 0.f};
#pragma unroll
  for (int t = 0; t < TSA_KNN; ++t) {
    const int ci = kidx[(size_t)wid * TSA_KNN + t];
    const float w = kw[(size_t)wid * TSA_KNN + t];
    const float* row = cf2 + (size_t)ci * TSA_DIM;
#pragma unroll
    for (int u = 0; u < 4; ++u) acc[u] += w * row[lane + u * 32];
  }
#pragma unroll
  for (int u = 0; u < 4; ++u)
    of[(size_t)wid * TSA_DIM + lane + u * 32] = acc[u];
}

// ---------------- Final: out = feats + of + relu(of @ Wp + bp) -------------
__global__ void tsa_final(const float* __restrict__ of,
                          const float* __restrict__ feats,
                          const _Float16* __restrict__ Wpp,
                          const float* __restrict__ bp, int rows,
                          float* __restrict__ out) {
  const int gw = (int)(blockIdx.x * blockDim.x + threadIdx.x) >> 5;
  const int ntile = TSA_DIM / 16, ktiles = TSA_DIM / 32;
  const int tiles_m = (rows + 15) >> 4;
  const int tm = gw / ntile, tn = gw - tm * ntile;
  if (tm >= tiles_m) return;
  const int lane = threadIdx.x & 31, l = lane & 15, hi = lane >> 4;
  const int row = tm * 16 + l;
  const bool av = row < rows;
  const int n0 = tn << 4;
  v8f acc = {};
  for (int kt = 0; kt < ktiles; ++kt) {
    v16h a = tsa_load_a(of, TSA_DIM, av ? row : 0, kt * 32 + hi * 8, av);
    v16h b = tsa_load_bp(Wpp, ktiles, tn, kt, lane);
    acc = __builtin_amdgcn_wmma_f32_16x16x32_f16(false, a, false, b, (short)0,
                                                 acc, false, false);
  }
  const float bb = bp[n0 + l];
#pragma unroll
  for (int i = 0; i < 8; ++i) {
    int r = tm * 16 + i + hi * 8;
    if (r < rows) {
      size_t o = (size_t)r * TSA_DIM + n0 + l;
      out[o] = feats[o] + of[o] + fmaxf(acc[i] + bb, 0.0f);
    }
  }
}

// ---------------------------------------------------------------------------
extern "C" void kernel_launch(void* const* d_in, const int* in_sizes, int n_in,
                              void* d_out, int out_size, void* d_ws,
                              size_t ws_size, hipStream_t stream) {
  const float* xyz   = (const float*)d_in[0];
  const float* feats = (const float*)d_in[1];
  const float* Wq    = (const float*)d_in[2];
  const float* Wk    = (const float*)d_in[3];
  const float* Wv    = (const float*)d_in[4];
  const float* Wo    = (const float*)d_in[5];
  const float* bo    = (const float*)d_in[6];
  const float* n1w   = (const float*)d_in[7];
  const float* n1b   = (const float*)d_in[8];
  const float* n2w   = (const float*)d_in[9];
  const float* n2b   = (const float*)d_in[10];
  const float* W1    = (const float*)d_in[11];
  const float* b1    = (const float*)d_in[12];
  const float* W2    = (const float*)d_in[13];
  const float* b2    = (const float*)d_in[14];
  const float* Wp    = (const float*)d_in[15];
  const float* bp    = (const float*)d_in[16];
  const int*   idxc  = (const int*)d_in[17];
  const int N = in_sizes[0] / 3;
  const int M = in_sizes[17];
  float* out = (float*)d_out;

  // workspace carve-out (256B aligned)
  char* base = (char*)d_ws;
  size_t off = 0;
  auto alloc = [&](size_t bytes) -> void* {
    void* p = base + off;
    off += (bytes + 255) & ~(size_t)255;
    return p;
  };
  int*      nbr  = (int*)alloc((size_t)M * TSA_MAXK * 4);
  float*    qm   = (float*)alloc((size_t)M * TSA_DIM * 4);
  _Float16* km   = (_Float16*)alloc((size_t)M * TSA_MAXK * TSA_DIM * 2);
  _Float16* vm   = (_Float16*)alloc((size_t)M * TSA_MAXK * TSA_DIM * 2);
  float*    outm = (float*)alloc((size_t)M * TSA_DIM * 4);
  float*    upd  = (float*)alloc((size_t)M * TSA_DIM * 4);
  float*    cf   = (float*)alloc((size_t)M * TSA_DIM * 4);
  float*    h1   = (float*)alloc((size_t)M * 4 * TSA_DIM * 4);
  float*    ffn  = (float*)alloc((size_t)M * TSA_DIM * 4);
  float*    cf2  = (float*)alloc((size_t)M * TSA_DIM * 4);
  int*      kidx = (int*)alloc((size_t)N * TSA_KNN * 4);
  float*    kw   = (float*)alloc((size_t)N * TSA_KNN * 4);
  float*    of   = (float*)alloc((size_t)N * TSA_DIM * 4);
  // pre-packed f16 weight fragments (B-layout, one v16h per (tile, lane))
  _Float16* Wqp = (_Float16*)alloc((size_t)TSA_DIM * TSA_DIM * 2);
  _Float16* Wkp = (_Float16*)alloc((size_t)TSA_DIM * TSA_DIM * 2);
  _Float16* Wvp = (_Float16*)alloc((size_t)TSA_DIM * TSA_DIM * 2);
  _Float16* Wop = (_Float16*)alloc((size_t)TSA_DIM * TSA_DIM * 2);
  _Float16* W1p = (_Float16*)alloc((size_t)TSA_DIM * 4 * TSA_DIM * 2);
  _Float16* W2p = (_Float16*)alloc((size_t)TSA_DIM * 4 * TSA_DIM * 2);
  _Float16* Wpp = (_Float16*)alloc((size_t)TSA_DIM * TSA_DIM * 2);

  const int tilesM  = (M + 15) / 16;
  const int rowsKV  = M * TSA_MAXK;
  const int tilesKV = (rowsKV + 15) / 16;
  const int tilesN  = (N + 15) / 16;
  const int nt128 = TSA_DIM / 16;       // 8
  const int nt512 = (4 * TSA_DIM) / 16; // 32
  auto blocks_for = [](int waves) { return (waves + 7) / 8; }; // 256 thr = 8 waves
  auto pack_blocks = [](int kdim, int ncols) {
    return ((ncols / 16) * (kdim / 32) * 32 + 255) / 256;
  };

  // 0. pre-pack all weights into WMMA B-fragment layout (one shot, tiny)
  tsa_pack_b<<<pack_blocks(TSA_DIM, TSA_DIM), 256, 0, stream>>>(Wq, TSA_DIM, TSA_DIM, Wqp);
  tsa_pack_b<<<pack_blocks(TSA_DIM, TSA_DIM), 256, 0, stream>>>(Wk, TSA_DIM, TSA_DIM, Wkp);
  tsa_pack_b<<<pack_blocks(TSA_DIM, TSA_DIM), 256, 0, stream>>>(Wv, TSA_DIM, TSA_DIM, Wvp);
  tsa_pack_b<<<pack_blocks(TSA_DIM, TSA_DIM), 256, 0, stream>>>(Wo, TSA_DIM, TSA_DIM, Wop);
  tsa_pack_b<<<pack_blocks(TSA_DIM, 4 * TSA_DIM), 256, 0, stream>>>(W1, TSA_DIM, 4 * TSA_DIM, W1p);
  tsa_pack_b<<<pack_blocks(4 * TSA_DIM, TSA_DIM), 256, 0, stream>>>(W2, 4 * TSA_DIM, TSA_DIM, W2p);
  tsa_pack_b<<<pack_blocks(TSA_DIM, TSA_DIM), 256, 0, stream>>>(Wp, TSA_DIM, TSA_DIM, Wpp);

  // 1. ball query -> nbr_idx (-1 = invalid)
  tsa_ballquery<<<M, 256, 0, stream>>>(xyz, idxc, N, nbr);
  // 2. q projection (gather fused)
  tsa_proj_center<<<blocks_for(tilesM * nt128), 256, 0, stream>>>(feats, idxc, M, Wqp, qm);
  // 3. k/v projections (neighbor gather fused, f16 output)
  tsa_proj_kv<<<blocks_for(tilesKV * nt128), 256, 0, stream>>>(feats, nbr, rowsKV, Wkp, Wvp, km, vm);
  // 4. masked attention + (D,H) flatten
  tsa_attn<<<M, TSA_DIM, 0, stream>>>(qm, km, vm, nbr, outm);
  // 5. upd = outm @ Wo + bo
  tsa_gemm_rm<<<blocks_for(tilesM * nt128), 256, 0, stream>>>(outm, M, TSA_DIM, Wop, TSA_DIM, bo, 0, upd);
  // 6. cf = center_feat + LN(upd)
  tsa_ln<<<M, TSA_DIM, 0, stream>>>(upd, feats, idxc, n1w, n1b, cf);
  // 7. h1 = relu(cf @ W1 + b1)
  tsa_gemm_rm<<<blocks_for(tilesM * nt512), 256, 0, stream>>>(cf, M, TSA_DIM, W1p, 4 * TSA_DIM, b1, 1, h1);
  // 8. ffn = h1 @ W2 + b2
  tsa_gemm_rm<<<blocks_for(tilesM * nt128), 256, 0, stream>>>(h1, M, 4 * TSA_DIM, W2p, TSA_DIM, b2, 0, ffn);
  // 9. cf2 = cf + LN(ffn)
  tsa_ln<<<M, TSA_DIM, 0, stream>>>(ffn, cf, (const int*)nullptr, n2w, n2b, cf2);
  // 10. KNN top-8 + normalized IDW weights
  tsa_knn<<<(N + 255) / 256, 256, 0, stream>>>(xyz, idxc, N, M, kidx, kw);
  // 11. of = IDW upsample of cf2 (one wave per point)
  tsa_upsample<<<(N * 32 + 255) / 256, 256, 0, stream>>>(cf2, kidx, kw, N, of);
  // 12. out = feats + of + relu(of @ Wp + bp)
  tsa_final<<<blocks_for(tilesN * nt128), 256, 0, stream>>>(of, feats, Wpp, bp, N, out);
}
